// GraphClassifier_64716567216697
// MI455X (gfx1250) — compile-verified, matrix operand order checked by reading, not run on hardware
//
#include <hip/hip_runtime.h>

// ---------------------------------------------------------------------------
// CDNA5 (gfx1250) wave32 WMMA types
// ---------------------------------------------------------------------------
typedef __attribute__((ext_vector_type(16))) __bf16 v16bf;
typedef __attribute__((ext_vector_type(8)))  float  v8f;
typedef __attribute__((ext_vector_type(4)))  float  v4f;   // for NT loads
typedef __attribute__((ext_vector_type(2)))  __bf16 v2bf;

#define NODES 4096

// fp32 pair -> packed bf16 dword: one v_cvt_pk_bf16_f32 per dword.
__device__ __forceinline__ unsigned int cvt2(float x, float y) {
#if __has_builtin(__builtin_amdgcn_cvt_pk_bf16_f32)
    v2bf p = __builtin_amdgcn_cvt_pk_bf16_f32(x, y);
    return __builtin_bit_cast(unsigned int, p);
#else
    union { __bf16 h[2]; unsigned int u; } r;
    r.h[0] = (__bf16)x; r.h[1] = (__bf16)y;
    return r.u;
#endif
}

__device__ __forceinline__ uint4 pack8(float4 a, float4 b) {
    uint4 r;
    r.x = cvt2(a.x, a.y);
    r.y = cvt2(a.z, a.w);
    r.z = cvt2(b.x, b.y);
    r.w = cvt2(b.z, b.w);
    return r;
}

// ---------------------------------------------------------------------------
// Tiled GEMM:  C[M][N] (+bias) = A[M][K] x Bt[N][K]^T   (Bt is "weights" layout)
// AMODE 0: A = fp32 activations.  AMODE 1: A[i][k] = adj[i][k]*alpha[i][k].
// bf16 operands staged through LDS, fp32 WMMA accumulation.
// Ping-pong LDS double buffer: tile k+1 global loads issue before tile k's
// WMMAs; one s_barrier per k-step.  global_prefetch two tiles ahead.
// Split-K via gridDim.z: block z writes partial C at offset z*M*N (bias=null).
// Block tile 128M x 64N x 32K; 8 wave32 waves in a 4(M) x 2(N) grid;
// each wave owns a 32x32 tile = 2x2 v_wmma_f32_16x16x32_bf16 accumulators.
// ---------------------------------------------------------------------------
template<int AMODE>
__device__ __forceinline__ void load_a_regs(const float* __restrict__ A,
                                            const int*   __restrict__ Aadj,
                                            const float* __restrict__ Aalpha,
                                            long base, float4 f[4])
{
    if constexpr (AMODE == 0) {
        const float4* p = reinterpret_cast<const float4*>(A + base);
        f[0] = p[0]; f[1] = p[1]; f[2] = p[2]; f[3] = p[3];
    } else {
        const int4* ip = reinterpret_cast<const int4*>(Aadj + base);
        const v4f*  ap = reinterpret_cast<const v4f*>(Aalpha + base);
        int4 i0 = ip[0], i1 = ip[1], i2 = ip[2], i3 = ip[3];
        // alpha is single-use streamed data: NT hint keeps x/adj resident in L2
        v4f a0 = __builtin_nontemporal_load(ap + 0);
        v4f a1 = __builtin_nontemporal_load(ap + 1);
        v4f a2 = __builtin_nontemporal_load(ap + 2);
        v4f a3 = __builtin_nontemporal_load(ap + 3);
        f[0] = make_float4(a0.x*i0.x, a0.y*i0.y, a0.z*i0.z, a0.w*i0.w);
        f[1] = make_float4(a1.x*i1.x, a1.y*i1.y, a1.z*i1.z, a1.w*i1.w);
        f[2] = make_float4(a2.x*i2.x, a2.y*i2.y, a2.z*i2.z, a2.w*i2.w);
        f[3] = make_float4(a3.x*i3.x, a3.y*i3.y, a3.z*i3.z, a3.w*i3.w);
    }
}

template<int AMODE>
__global__ __launch_bounds__(256)
void gemm_bf16_wmma(int M, int N, int K,
                    const float* __restrict__ A,
                    const int*   __restrict__ Aadj,
                    const float* __restrict__ Aalpha,
                    const float* __restrict__ Bt,
                    const float* __restrict__ bias,
                    float* __restrict__ C)
{
    // 48-half row stride keeps every 8-half chunk 16B-aligned (96B rows)
    __shared__ __align__(16) unsigned short As[2][128][48];
    __shared__ __align__(16) unsigned short Bs[2][64][48];

    const int tid  = threadIdx.x;
    const int lane = tid & 31;
    const int wave = tid >> 5;
    const int wm   = wave & 3;       // 4 waves along M (32 rows each)
    const int wn   = wave >> 2;      // 2 waves along N (32 cols each)
    const int lo   = lane & 15;
    const int hi   = lane >> 4;      // half-wave select

    const int n0 = blockIdx.x * 64;
    const int m0 = blockIdx.y * 128;

    // split-K range for this z-slice
    const int kPer = K / gridDim.z;
    const int kb   = blockIdx.z * kPer;
    const int ke   = kb + kPer;

    // staging assignments: A tile 128x32 (16 halfs/thread), B tile 64x32 (8/thread)
    const int arow = tid >> 1;
    const int acol = (tid & 1) * 16;
    const int brow = tid >> 2;
    const int bcol = (tid & 3) * 8;

    const long aBase = (long)(m0 + arow) * K + acol;
    const long bBase = (long)(n0 + brow) * K + bcol;

    v8f acc[2][2] = {};

    // ---- prologue: stage tile kb into buffer 0 ----
    {
        float4 fa[4];
        load_a_regs<AMODE>(A, Aadj, Aalpha, aBase + kb, fa);
        const float4* pb = reinterpret_cast<const float4*>(Bt + bBase + kb);
        float4 g0 = pb[0], g1 = pb[1];
        uint4* da = reinterpret_cast<uint4*>(&As[0][arow][acol]);
        da[0] = pack8(fa[0], fa[1]);
        da[1] = pack8(fa[2], fa[3]);
        *reinterpret_cast<uint4*>(&Bs[0][brow][bcol]) = pack8(g0, g1);
    }
    __syncthreads();

    int cur = 0;
    for (int k0 = kb; k0 < ke; k0 += 32) {
        const bool hasNext = (k0 + 32) < ke;

        // ---- issue next tile's global loads before this tile's math ----
        float4 fa[4], g0, g1;
        if (hasNext) {
            load_a_regs<AMODE>(A, Aadj, Aalpha, aBase + k0 + 32, fa);
            const float4* pb = reinterpret_cast<const float4*>(Bt + bBase + k0 + 32);
            g0 = pb[0]; g1 = pb[1];
        }
        // ---- prefetch two tiles ahead (global_prefetch_b8) ----
        if (k0 + 64 < ke) {
            if constexpr (AMODE == 0) {
                __builtin_prefetch(A + aBase + k0 + 64, 0, 3);
            } else {
                __builtin_prefetch(Aadj   + aBase + k0 + 64, 0, 3);
                __builtin_prefetch(Aalpha + aBase + k0 + 64, 0, 3);
            }
            __builtin_prefetch(Bt + bBase + k0 + 64, 0, 3);
        }

        // ---- fragments per the 16-bit A-matrix 16x32 layout:
        //      lane<16: K = {0..7, 16..23}; lane>=16: K = {8..15, 24..31}
        union Frag { uint4 u[2]; v16bf v; };
        Frag a[2], b[2];
        #pragma unroll
        for (int mt = 0; mt < 2; ++mt) {
            const uint4* p = reinterpret_cast<const uint4*>(&As[cur][wm*32 + mt*16 + lo][0]);
            a[mt].u[0] = p[hi];
            a[mt].u[1] = p[hi + 2];
        }
        #pragma unroll
        for (int nt = 0; nt < 2; ++nt) {
            const uint4* p = reinterpret_cast<const uint4*>(&Bs[cur][wn*32 + nt*16 + lo][0]);
            b[nt].u[0] = p[hi];
            b[nt].u[1] = p[hi + 2];
        }
        #pragma unroll
        for (int mt = 0; mt < 2; ++mt)
            #pragma unroll
            for (int nt = 0; nt < 2; ++nt)
                acc[mt][nt] = __builtin_amdgcn_wmma_f32_16x16x32_bf16(
                    false, a[mt].v, false, b[nt].v,
                    (short)0, acc[mt][nt], false, false);

        // ---- convert + store next tile into the other buffer ----
        if (hasNext) {
            uint4* da = reinterpret_cast<uint4*>(&As[cur ^ 1][arow][acol]);
            da[0] = pack8(fa[0], fa[1]);
            da[1] = pack8(fa[2], fa[3]);
            *reinterpret_cast<uint4*>(&Bs[cur ^ 1][brow][bcol]) = pack8(g0, g1);
        }
        __syncthreads();
        cur ^= 1;
    }

    // ---- epilogue: C/D layout — lane<16: M=r, N=lane; lane>=16: M=8+r, N=lane-16
    float* Cz = C + (size_t)blockIdx.z * M * N;
    #pragma unroll
    for (int mt = 0; mt < 2; ++mt) {
        #pragma unroll
        for (int nt = 0; nt < 2; ++nt) {
            const int n  = n0 + wn*32 + nt*16 + lo;
            const float bv = bias ? bias[n] : 0.0f;
            #pragma unroll
            for (int r = 0; r < 8; ++r) {
                const int m = m0 + wm*32 + mt*16 + hi*8 + r;
                Cz[(size_t)m * N + n] = acc[mt][nt][r] + bv;
            }
        }
    }
}

// ---------------------------------------------------------------------------
// BatchNorm column stats over R rows: scale = g*rsqrt(var+eps), shift = be - m*scale
// Block covers 64 columns; 4 row-strips reduced through LDS.
// ---------------------------------------------------------------------------
__global__ __launch_bounds__(256)
void bn_stats(const float* __restrict__ X, int R, int C,
              const float* __restrict__ g, const float* __restrict__ be,
              float* __restrict__ scale, float* __restrict__ shift)
{
    __shared__ float s1[256], s2[256];
    const int t = threadIdx.x;
    const int c = blockIdx.x * 64 + (t & 63);
    const int q = t >> 6;
    float sum = 0.f, sq = 0.f;
    for (int r = q; r < R; r += 4) {
        float x = X[(size_t)r * C + c];
        sum += x; sq += x * x;
    }
    s1[t] = sum; s2[t] = sq;
    __syncthreads();
    if (q == 0) {
        sum = s1[t] + s1[t+64] + s1[t+128] + s1[t+192];
        sq  = s2[t] + s2[t+64] + s2[t+128] + s2[t+192];
        float m  = sum / (float)R;
        float v  = sq / (float)R - m * m;
        float sc = g[c] * rsqrtf(v + 1e-5f);
        scale[c] = sc;
        shift[c] = be[c] - m * sc;
    }
}

__global__ void bn_apply_relu(float* __restrict__ X, int total, int cmask,
                              const float* __restrict__ scale,
                              const float* __restrict__ shift)
{
    int i = blockIdx.x * blockDim.x + threadIdx.x;
    int stride = gridDim.x * blockDim.x;
    for (; i < total; i += stride) {
        int c = i & cmask;
        X[i] = fmaxf(0.f, fmaf(X[i], scale[c], shift[c]));
    }
}

// h [4096][64] -> hT [64][4096] via LDS 32x32 tiles (both sides coalesced)
__global__ __launch_bounds__(256)
void transpose_h(const float* __restrict__ H, float* __restrict__ HT)
{
    __shared__ float tile[32][33];
    const int t  = threadIdx.x;
    const int tx = t & 31, ty = t >> 5;
    const int cb = blockIdx.x * 32;
    const int rb = blockIdx.y * 32;
    #pragma unroll
    for (int i = 0; i < 32; i += 8)
        tile[ty + i][tx] = H[(size_t)(rb + ty + i) * 64 + cb + tx];
    __syncthreads();
    #pragma unroll
    for (int i = 0; i < 32; i += 8)
        HT[(size_t)(cb + ty + i) * NODES + rb + tx] = tile[tx][ty + i];
}

// invdeg[row] = W / sum_j adj[row][j]  (also warms adj into the 192MB L2
// right before the aggregation GEMM streams it again)
__global__ __launch_bounds__(256)
void row_invdeg(const int* __restrict__ adj, const float* __restrict__ Wptr,
                float* __restrict__ invdeg)
{
    __shared__ int sd[256];
    const int row = blockIdx.x;
    const int t = threadIdx.x;
    const int4* p = reinterpret_cast<const int4*>(adj + (size_t)row * NODES);
    int acc = 0;
    for (int i = t; i < NODES / 4; i += 256) {
        int4 v = p[i];
        acc += v.x + v.y + v.z + v.w;
    }
    sd[t] = acc; __syncthreads();
    for (int s = 128; s > 0; s >>= 1) {
        if (t < s) sd[t] += sd[t + s];
        __syncthreads();
    }
    if (t == 0) invdeg[row] = Wptr[0] / (float)sd[0];
}

// reduce 8 split-K partials and apply per-row W/deg scaling -> feat slice
__global__ void agg_reduce_scale(const float* __restrict__ part,
                                 const float* __restrict__ invdeg,
                                 float* __restrict__ feat)
{
    const int total = NODES * 64;
    int i = blockIdx.x * blockDim.x + threadIdx.x;
    int stride = gridDim.x * blockDim.x;
    for (; i < total; i += stride) {
        float s = 0.f;
        #pragma unroll
        for (int z = 0; z < 8; ++z) s += part[(size_t)z * total + i];
        feat[i] = s * invdeg[i >> 6];
    }
}

// classifier: two L-length dot products, deterministic two-stage reduction
__global__ __launch_bounds__(256)
void cls_dot_partial(const float* __restrict__ feat, const float* __restrict__ W,
                     int L, float* __restrict__ partials)
{
    __shared__ float s0[256], s1[256];
    const int t = threadIdx.x;
    float p0 = 0.f, p1 = 0.f;
    for (int i = blockIdx.x * 256 + t; i < L; i += gridDim.x * 256) {
        float f = feat[i];
        p0 = fmaf(f, W[i],     p0);
        p1 = fmaf(f, W[L + i], p1);
    }
    s0[t] = p0; s1[t] = p1; __syncthreads();
    for (int s = 128; s > 0; s >>= 1) {
        if (t < s) { s0[t] += s0[t + s]; s1[t] += s1[t + s]; }
        __syncthreads();
    }
    if (t == 0) { partials[blockIdx.x * 2] = s0[0]; partials[blockIdx.x * 2 + 1] = s1[0]; }
}

__global__ __launch_bounds__(256)
void cls_finalize(const float* __restrict__ partials, int nPart,
                  const float* __restrict__ cls_b, float* __restrict__ out)
{
    __shared__ float s0[256], s1[256];
    const int t = threadIdx.x;
    float p0 = 0.f, p1 = 0.f;
    for (int i = t; i < nPart; i += 256) { p0 += partials[i*2]; p1 += partials[i*2+1]; }
    s0[t] = p0; s1[t] = p1; __syncthreads();
    for (int s = 128; s > 0; s >>= 1) {
        if (t < s) { s0[t] += s0[t + s]; s1[t] += s1[t + s]; }
        __syncthreads();
    }
    if (t == 0) {
        float l0 = s0[0] + cls_b[0], l1 = s1[0] + cls_b[1];
        float m  = fmaxf(l0, l1);
        float e0 = __expf(l0 - m), e1 = __expf(l1 - m);
        float inv = 1.0f / (e0 + e1);
        out[0] = e0 * inv; out[1] = e1 * inv;
    }
}

// ---------------------------------------------------------------------------
// Orchestration.  Input flat order (setup_inputs dict, recursive):
//  0:x1 1:x2 2:adj1 3:adj2
//  4..15:  enc1 {W0,b0,g0,be0, W1,b1,g1,be1, W2,b2,g2,be2}
//  16..27: enc2 (same layout)
//  28:W(1,1) 29:alpha1 30:alpha2 31:cls_W 32:cls_b
// ---------------------------------------------------------------------------
extern "C" void kernel_launch(void* const* d_in, const int* in_sizes, int n_in,
                              void* d_out, int out_size, void* d_ws, size_t ws_size,
                              hipStream_t stream)
{
    (void)in_sizes; (void)n_in; (void)out_size; (void)ws_size;

    const float* x[2]     = { (const float*)d_in[0], (const float*)d_in[1] };
    const int*   adj[2]   = { (const int*)d_in[2],   (const int*)d_in[3] };
    const float* Wsc      = (const float*)d_in[28];
    const float* alpha[2] = { (const float*)d_in[29], (const float*)d_in[30] };
    const float* clsW     = (const float*)d_in[31];
    const float* clsb     = (const float*)d_in[32];
    float*       out      = (float*)d_out;

    // workspace carve-up (floats)
    float* ws = (float*)d_ws;
    size_t o = 0;
    float* feat   = ws + o; o += 2 * NODES * 64;       // 524288
    float* act0   = ws + o; o += (size_t)NODES * 256;
    float* act1   = ws + o; o += (size_t)NODES * 128;
    float* act2   = ws + o; o += (size_t)NODES * 64;
    float* hT     = ws + o; o += (size_t)64 * NODES;
    float* aggp   = ws + o; o += (size_t)8 * NODES * 64;  // split-K partials
    float* scale  = ws + o; o += 256;
    float* shift  = ws + o; o += 256;
    float* invdeg = ws + o; o += NODES;
    float* parts  = ws + o; o += 512;

    const int M = NODES;

    for (int g = 0; g < 2; ++g) {
        void* const* P = d_in + (g ? 16 : 4);
        const float* W0 = (const float*)P[0],  *b0 = (const float*)P[1];
        const float* g0 = (const float*)P[2],  *be0 = (const float*)P[3];
        const float* W1 = (const float*)P[4],  *b1 = (const float*)P[5];
        const float* g1 = (const float*)P[6],  *be1 = (const float*)P[7];
        const float* W2 = (const float*)P[8],  *b2 = (const float*)P[9];
        const float* g2 = (const float*)P[10], *be2 = (const float*)P[11];

        // Layer 0: [4096x4096]·[4096x256]  (x lives in L2 across N-blocks)
        gemm_bf16_wmma<0><<<dim3(4, M/128, 1), 256, 0, stream>>>(
            M, 256, 4096, x[g], nullptr, nullptr, W0, b0, act0);
        bn_stats<<<dim3(4), 256, 0, stream>>>(act0, M, 256, g0, be0, scale, shift);
        bn_apply_relu<<<dim3(1024), 256, 0, stream>>>(act0, M*256, 255, scale, shift);

        // Layer 1: [4096x256]·[256x128]
        gemm_bf16_wmma<0><<<dim3(2, M/128, 1), 256, 0, stream>>>(
            M, 128, 256, act0, nullptr, nullptr, W1, b1, act1);
        bn_stats<<<dim3(2), 256, 0, stream>>>(act1, M, 128, g1, be1, scale, shift);
        bn_apply_relu<<<dim3(512), 256, 0, stream>>>(act1, M*128, 127, scale, shift);

        // Layer 2: [4096x128]·[128x64] -> h
        gemm_bf16_wmma<0><<<dim3(1, M/128, 1), 256, 0, stream>>>(
            M, 64, 128, act1, nullptr, nullptr, W2, b2, act2);
        bn_stats<<<dim3(1), 256, 0, stream>>>(act2, M, 64, g2, be2, scale, shift);
        bn_apply_relu<<<dim3(256), 256, 0, stream>>>(act2, M*64, 63, scale, shift);

        // h^T for weights-layout B operand of the aggregation GEMM
        transpose_h<<<dim3(2, 128), 256, 0, stream>>>(act2, hT);

        // deg first: streams adj through L2 so the GEMM re-read hits cache
        row_invdeg<<<dim3(NODES), 256, 0, stream>>>(adj[g], Wsc, invdeg);

        // aggregation: (adj⊙alpha) @ h, split-K=8 for occupancy (N=64 only)
        gemm_bf16_wmma<1><<<dim3(1, M/128, 8), 256, 0, stream>>>(
            M, 64, 4096, nullptr, adj[g], alpha[g], hT, nullptr, aggp);
        agg_reduce_scale<<<dim3(1024), 256, 0, stream>>>(aggp, invdeg, feat + (size_t)g * NODES * 64);
    }

    // classifier + softmax
    cls_dot_partial<<<dim3(256), 256, 0, stream>>>(feat, clsW, 2 * NODES * 64, parts);
    cls_finalize<<<dim3(1), 256, 0, stream>>>(parts, 256, clsb, out);
}
